// TaskAlignedAssigner_40037685133862
// MI455X (gfx1250) — matrix operand order repaired and checked by reading, not compiled, not run on hardware
//
#include <hip/hip_runtime.h>
#include <hip/hip_bf16.h>

#define BS    16
#define NA    8400
#define NMB   100
#define NC    80
#define TOPK  13
#define EPSF  1e-9f

typedef __attribute__((ext_vector_type(16))) _Float16 v16h;
typedef __attribute__((ext_vector_type(8)))  float    v8f;
typedef __attribute__((ext_vector_type(4)))  float    v4f;

// ---------------------------------------------------------------------------
// Kernel 1: align metric via WMMA one-hot gather + fused IoU.
// 256-thread block = 8 waves; each wave computes one 16(gt) x 16(anchor) tile:
//   bbox_scores = onehot(labels)[16x96] @ scores^T[96x16]  (3 x wmma K=32)
//   metric = score * iou^6 ; stored as -1 sentinel when anchor not in gt box.
// ---------------------------------------------------------------------------
__global__ __launch_bounds__(256) void k1_metric(
    const float* __restrict__ scores,   // [BS,NA,NC]
    const float* __restrict__ pbox,     // [BS,NA,4]
    const float* __restrict__ anch,     // [NA,2]
    const float* __restrict__ gbox,     // [BS,NMB,4]
    const int*   __restrict__ glab,     // [BS,NMB]
    float* __restrict__ metric)         // [BS,NMB,NA]
{
    const int b    = blockIdx.z;
    const int m0   = blockIdx.y * 16;
    const int wave = threadIdx.x >> 5;
    const int lane = threadIdx.x & 31;
    const int half = lane >> 4;
    const int lm   = lane & 15;
    const int a0   = (blockIdx.x * 8 + wave) * 16;

    __shared__ float sg[16][4];
    __shared__ int   sl[16];
    if (threadIdx.x < 16) {
        int m = m0 + threadIdx.x;
        if (m < NMB) {
            const float* g = gbox + ((size_t)b * NMB + m) * 4;
            sg[threadIdx.x][0] = g[0]; sg[threadIdx.x][1] = g[1];
            sg[threadIdx.x][2] = g[2]; sg[threadIdx.x][3] = g[3];
            sl[threadIdx.x] = glab[b * NMB + m];
        } else {
            sg[threadIdx.x][0] = sg[threadIdx.x][1] = 0.f;
            sg[threadIdx.x][2] = sg[threadIdx.x][3] = 0.f;
            sl[threadIdx.x] = -1;
        }
    }
    __syncthreads();
    if (a0 >= NA) return;                       // tail waves (525 = 65*8 + 5)

    const int   labA = sl[lm];                  // A-matrix row (gt) for this lane
    const int   aN   = a0 + lm;                 // B/D column (anchor)
    const float* srow = scores + ((size_t)b * NA + aN) * NC;

    v8f c = {};
    #pragma unroll
    for (int t = 0; t < 3; ++t) {
        const int k0  = t * 32;
        const int kb0 = k0 + 16 * half;         // first K this lane supplies
        v16h A, B = {};
        if (kb0 < NC) {                         // uniform per lane-half/tile
            const v4f* sp = (const v4f*)(srow + kb0);   // 64B-aligned
            v4f q0 = sp[0], q1 = sp[1], q2 = sp[2], q3 = sp[3];
            B[0]  = (_Float16)q0.x; B[1]  = (_Float16)q0.y;
            B[2]  = (_Float16)q0.z; B[3]  = (_Float16)q0.w;
            B[4]  = (_Float16)q1.x; B[5]  = (_Float16)q1.y;
            B[6]  = (_Float16)q1.z; B[7]  = (_Float16)q1.w;
            B[8]  = (_Float16)q2.x; B[9]  = (_Float16)q2.y;
            B[10] = (_Float16)q2.z; B[11] = (_Float16)q2.w;
            B[12] = (_Float16)q3.x; B[13] = (_Float16)q3.y;
            B[14] = (_Float16)q3.z; B[15] = (_Float16)q3.w;
        }
        #pragma unroll
        for (int e = 0; e < 16; ++e) {
            // A order used here (contiguous, same as B): lane -> M=lane%16; K = e + 16*half
            int ka = k0 + e + 16 * half;
            A[e] = (labA == ka) ? (_Float16)1.0f : (_Float16)0.0f;
        }
        c = __builtin_amdgcn_wmma_f32_16x16x32_f16(false, A, false, B,
                                                   (short)0, c, false, false);
    }

    // Per-lane anchor data
    const v4f pbv = *(const v4f*)(pbox + ((size_t)b * NA + aN) * 4);
    float px0 = pbv.x, px1 = pbv.y, px2 = pbv.z, px3 = pbv.w;
    float ax = anch[aN * 2 + 0], ay = anch[aN * 2 + 1];
    float areap = (px2 - px0) * (px3 - px1);

    #pragma unroll
    for (int j = 0; j < 8; ++j) {
        int mr = j + 8 * half;          // D layout: VGPR j -> M = j + 8*half
        int m  = m0 + mr;
        if (m >= NMB) continue;
        float g0 = sg[mr][0], g1 = sg[mr][1], g2 = sg[mr][2], g3 = sg[mr][3];
        float iw = fminf(g2, px2) - fmaxf(g0, px0);
        float ih = fminf(g3, px3) - fmaxf(g1, px1);
        float inter = fmaxf(iw, 0.f) * fmaxf(ih, 0.f);
        float areag = (g2 - g0) * (g3 - g1);
        float iou = inter / (areag + areap - inter + EPSF);
        float dmin = fminf(fminf(ax - g0, ay - g1), fminf(g2 - ax, g3 - ay));
        float iou2 = iou * iou;
        float met  = c[j] * iou2 * iou2 * iou2;          // score^1 * iou^6
        metric[((size_t)(b * NMB + m)) * NA + aN] = (dmin > EPSF) ? met : -1.0f;
    }
}

// ---------------------------------------------------------------------------
// Kernel 2: exact top-13 per (b,m); tie-break = lowest index (matches lax.top_k)
// ---------------------------------------------------------------------------
__global__ __launch_bounds__(256) void k2_topk(
    const float* __restrict__ metric, const float* __restrict__ mask_gt,
    unsigned char* __restrict__ maskpos)
{
    const int m = blockIdx.x, b = blockIdx.y, tid = threadIdx.x;
    const size_t row = ((size_t)(b * NMB + m)) * NA;

    if (mask_gt[b * NMB + m] <= 0.f) {
        // reference: topk idxs forced to 0 -> count 13 > 1 -> all dropped
        for (int a = tid; a < NA; a += 256) maskpos[row + a] = 0;
        return;
    }

    __shared__ float         sval[NA];
    __shared__ unsigned char ssel[NA];
    __shared__ float rv[256];
    __shared__ int   ri[256];

    for (int a = tid; a < NA; a += 256) {
        __builtin_prefetch(&metric[row + a + 2048], 0, 0);
        sval[a] = metric[row + a];
        ssel[a] = 0;
    }
    __syncthreads();

    for (int t = 0; t < TOPK; ++t) {
        float bv = -1.f; int bi = NA;
        for (int a = tid; a < NA; a += 256) {
            if (ssel[a]) continue;
            float v = fmaxf(sval[a], 0.f);   // sentinel (-1) ranks as 0, like ref
            if (v > bv || (v == bv && a < bi)) { bv = v; bi = a; }
        }
        rv[tid] = bv; ri[tid] = bi;
        __syncthreads();
        for (int s = 128; s > 0; s >>= 1) {
            if (tid < s) {
                float ov = rv[tid + s]; int oi = ri[tid + s];
                if (ov > rv[tid] || (ov == rv[tid] && oi < ri[tid])) {
                    rv[tid] = ov; ri[tid] = oi;
                }
            }
            __syncthreads();
        }
        if (tid == 0) ssel[ri[0]] = 1;
        __syncthreads();
    }

    for (int a = tid; a < NA; a += 256)
        maskpos[row + a] = (ssel[a] && sval[a] >= 0.f) ? 1 : 0;   // * is_in_gts
}

// ---------------------------------------------------------------------------
// Kernel 3: per-anchor multi-GT collision resolution + label/bbox/fg outputs
// ---------------------------------------------------------------------------
__global__ __launch_bounds__(256) void k3_resolve(
    const float* __restrict__ scores, const float* __restrict__ pbox,
    const float* __restrict__ gbox,   const int* __restrict__ glab,
    unsigned char* __restrict__ maskpos, float* __restrict__ metric,
    float* __restrict__ out_lab, float* __restrict__ out_box,
    float* __restrict__ out_fg)
{
    const int b = blockIdx.y;
    const int a = blockIdx.x * 256 + threadIdx.x;

    __shared__ float sg[NMB][4];
    __shared__ int   sl[NMB];
    for (int m = threadIdx.x; m < NMB; m += 256) {
        const float* g = gbox + ((size_t)b * NMB + m) * 4;
        sg[m][0] = g[0]; sg[m][1] = g[1]; sg[m][2] = g[2]; sg[m][3] = g[3];
        sl[m] = glab[b * NMB + m];
    }
    __syncthreads();
    if (a >= NA) return;

    const size_t base = (size_t)b * NMB * NA + a;
    int sum = 0, first = -1;
    for (int m = 0; m < NMB; ++m) {
        int v = maskpos[base + (size_t)m * NA];
        sum += v;
        if (v && first < 0) first = m;
    }

    int idx;
    if (sum > 1) {
        const v4f pbv = *(const v4f*)(pbox + ((size_t)b * NA + a) * 4);
        float px0 = pbv.x, px1 = pbv.y, px2 = pbv.z, px3 = pbv.w;
        float areap = (px2 - px0) * (px3 - px1);
        float best = -1.f; int bm = 0;
        for (int m = 0; m < NMB; ++m) {           // argmax over ALL gts, first max
            float g0 = sg[m][0], g1 = sg[m][1], g2 = sg[m][2], g3 = sg[m][3];
            float iw = fminf(g2, px2) - fmaxf(g0, px0);
            float ih = fminf(g3, px3) - fmaxf(g1, px1);
            float inter = fmaxf(iw, 0.f) * fmaxf(ih, 0.f);
            float areag = (g2 - g0) * (g3 - g1);
            float iou = inter / (areag + areap - inter + EPSF);
            if (iou > best) { best = iou; bm = m; }
        }
        idx = bm;
        for (int m = 0; m < NMB; ++m)
            maskpos[base + (size_t)m * NA] = (unsigned char)(m == bm);
        // winning gt may have in_gts==0 here -> stored metric is sentinel;
        // reference normalization uses the ungated metric, so recompute it.
        size_t mi = base + (size_t)bm * NA;
        if (metric[mi] < 0.f) {
            float sc = scores[((size_t)b * NA + a) * NC + sl[bm]];
            float i2 = best * best;
            metric[mi] = sc * i2 * i2 * i2;
        }
    } else {
        idx = (sum == 1) ? first : 0;
    }

    int lab = sl[idx]; if (lab < 0) lab = 0;
    size_t o = (size_t)b * NA + a;
    out_lab[o] = (float)lab;
    v4f bx = { sg[idx][0], sg[idx][1], sg[idx][2], sg[idx][3] };
    *(v4f*)(out_box + o * 4) = bx;                       // single b128 store
    out_fg[o] = (sum > 0) ? 1.f : 0.f;
}

// ---------------------------------------------------------------------------
// Kernel 4: per-GT max of masked metric / masked IoU (IoU recomputed)
// ---------------------------------------------------------------------------
__global__ __launch_bounds__(256) void k4_posmax(
    const float* __restrict__ metric, const unsigned char* __restrict__ maskpos,
    const float* __restrict__ pbox,   const float* __restrict__ gbox,
    float* __restrict__ posA, float* __restrict__ posO)
{
    const int m = blockIdx.x, b = blockIdx.y, tid = threadIdx.x;
    const size_t row = ((size_t)(b * NMB + m)) * NA;
    const float* g = gbox + ((size_t)b * NMB + m) * 4;
    float g0 = g[0], g1 = g[1], g2 = g[2], g3 = g[3];
    float areag = (g2 - g0) * (g3 - g1);

    float ma = 0.f, mo = 0.f;
    for (int a = tid; a < NA; a += 256) {
        if (!maskpos[row + a]) continue;
        float v = metric[row + a];
        ma = fmaxf(ma, fmaxf(v, 0.f));
        const v4f pbv = *(const v4f*)(pbox + ((size_t)b * NA + a) * 4);
        float px0 = pbv.x, px1 = pbv.y, px2 = pbv.z, px3 = pbv.w;
        float iw = fminf(g2, px2) - fmaxf(g0, px0);
        float ih = fminf(g3, px3) - fmaxf(g1, px1);
        float inter = fmaxf(iw, 0.f) * fmaxf(ih, 0.f);
        float areap = (px2 - px0) * (px3 - px1);
        mo = fmaxf(mo, inter / (areag + areap - inter + EPSF));
    }

    __shared__ float ra[256], ro[256];
    ra[tid] = ma; ro[tid] = mo;
    __syncthreads();
    for (int s = 128; s > 0; s >>= 1) {
        if (tid < s) {
            ra[tid] = fmaxf(ra[tid], ra[tid + s]);
            ro[tid] = fmaxf(ro[tid], ro[tid + s]);
        }
        __syncthreads();
    }
    if (tid == 0) { posA[b * NMB + m] = ra[0]; posO[b * NMB + m] = ro[0]; }
}

// ---------------------------------------------------------------------------
// Kernel 5: per-anchor norm + one-hot score scatter (vectorized b128 stores)
// ---------------------------------------------------------------------------
__global__ __launch_bounds__(256) void k5_scores(
    const float* __restrict__ metric, const unsigned char* __restrict__ maskpos,
    const float* __restrict__ posA,   const float* __restrict__ posO,
    const float* __restrict__ out_lab, const float* __restrict__ out_fg,
    float* __restrict__ out_scores)
{
    const int b = blockIdx.y;
    const int a = blockIdx.x * 256 + threadIdx.x;

    __shared__ float sA[NMB], sO[NMB];
    for (int m = threadIdx.x; m < NMB; m += 256) {
        sA[m] = posA[b * NMB + m];
        sO[m] = posO[b * NMB + m];
    }
    __syncthreads();
    if (a >= NA) return;

    const size_t col = (size_t)b * NMB * NA + a;
    float norm = 0.f;
    for (int m = 0; m < NMB; ++m) {
        if (!maskpos[col + (size_t)m * NA]) continue;
        float v = fmaxf(metric[col + (size_t)m * NA], 0.f);
        norm = fmaxf(norm, v * sO[m] / (sA[m] + EPSF));
    }

    size_t o = (size_t)b * NA + a;
    float fg = out_fg[o];
    int lab = (int)out_lab[o];
    float* so = out_scores + o * NC;     // 320B-aligned row
    v4f z = {0.f, 0.f, 0.f, 0.f};
    #pragma unroll
    for (int c4 = 0; c4 < NC / 4; ++c4)
        ((v4f*)so)[c4] = z;              // 20 x global_store_b128
    if (fg > 0.f) so[lab] = norm;        // single scalar fix-up
}

// ---------------------------------------------------------------------------
extern "C" void kernel_launch(void* const* d_in, const int* in_sizes, int n_in,
                              void* d_out, int out_size, void* d_ws, size_t ws_size,
                              hipStream_t stream) {
    const float* pred_scores   = (const float*)d_in[0];
    const float* pred_bboxes   = (const float*)d_in[1];
    const float* anchor_points = (const float*)d_in[2];
    const float* gt_bboxes     = (const float*)d_in[3];
    const float* mask_gt       = (const float*)d_in[4];
    const int*   gt_labels     = (const int*)d_in[5];

    float* out     = (float*)d_out;
    float* out_lab = out;                           // [BS,NA]
    float* out_box = out + (size_t)BS * NA;         // [BS,NA,4]
    float* out_sc  = out + (size_t)BS * NA * 5;     // [BS,NA,80]
    float* out_fg  = out + (size_t)BS * NA * 85;    // [BS,NA]

    char* ws = (char*)d_ws;
    float*         metric  = (float*)ws;                                   // 53.76 MB
    unsigned char* maskpos = (unsigned char*)(ws + (size_t)BS*NMB*NA*4);   // 13.44 MB
    float*         posA    = (float*)(ws + (size_t)BS*NMB*NA*5);           // 6.4 KB
    float*         posO    = posA + BS * NMB;                              // 6.4 KB

    // 525 anchor tiles of 16, grouped 8 tiles (8 waves) per block
    k1_metric<<<dim3((525 + 7) / 8, (NMB + 15) / 16, BS), 256, 0, stream>>>(
        pred_scores, pred_bboxes, anchor_points, gt_bboxes, gt_labels, metric);

    k2_topk<<<dim3(NMB, BS), 256, 0, stream>>>(metric, mask_gt, maskpos);

    k3_resolve<<<dim3((NA + 255) / 256, BS), 256, 0, stream>>>(
        pred_scores, pred_bboxes, gt_bboxes, gt_labels, maskpos, metric,
        out_lab, out_box, out_fg);

    k4_posmax<<<dim3(NMB, BS), 256, 0, stream>>>(
        metric, maskpos, pred_bboxes, gt_bboxes, posA, posO);

    k5_scores<<<dim3((NA + 255) / 256, BS), 256, 0, stream>>>(
        metric, maskpos, posA, posO, out_lab, out_fg, out_sc);
}
